// TransformerBlock_10024453669675
// MI455X (gfx1250) — matrix operand order repaired and checked by reading, not compiled, not run on hardware
//
#include <hip/hip_runtime.h>
#include <hip/hip_bf16.h>
#include <math.h>

// ---------------------------------------------------------------- constants
#define BB 4
#define NN 2048
#define DD 2048
#define HH 16
#define EE 1024
#define DHH 128
#define MM (BB*NN)          // 8192 rows

typedef __bf16 bf16;
typedef __attribute__((ext_vector_type(16))) __bf16 v16bf;
typedef __attribute__((ext_vector_type(8)))  __bf16 v8bf;
typedef __attribute__((ext_vector_type(8)))  float  v8f;
typedef __attribute__((ext_vector_type(4)))  int    v4i;

union Frag { v16bf v; v8bf h[2]; };

__device__ __forceinline__ v8f wmma_bf16(const v16bf& a, const v16bf& b, const v8f& c) {
    return __builtin_amdgcn_wmma_f32_16x16x32_bf16(false, a, false, b, (short)0, c, false, false);
}

// --------------------------------------------------------------- async global->LDS (gfx1250)
#if defined(__gfx1250__) && __has_builtin(__builtin_amdgcn_global_load_async_to_lds_b128) && __has_builtin(__builtin_amdgcn_s_wait_asynccnt)
#define ASYNC_LDS 1
#else
#define ASYNC_LDS 0
#endif

// copy 8 bf16 (16 bytes) global -> LDS
__device__ __forceinline__ void cp8(bf16* l, const bf16* g) {
#if ASYNC_LDS
    __builtin_amdgcn_global_load_async_to_lds_b128(
        (__attribute__((address_space(1))) v4i*)g,
        (__attribute__((address_space(3))) v4i*)l, 0, 0);
#else
    *(v8bf*)l = *(const v8bf*)g;
#endif
}
__device__ __forceinline__ void wait_async() {
#if ASYNC_LDS
    __builtin_amdgcn_s_wait_asynccnt(0);
#endif
}

// ---------------------------------------------------------------- casts
__global__ void k_cast_bf16(const float* __restrict__ in, bf16* __restrict__ out, long n) {
    long i = (long)blockIdx.x * blockDim.x + threadIdx.x;
    if (i < n) out[i] = (bf16)in[i];
}

// W (K x Nn) f32 row-major  ->  Wt (Nn x K) bf16 row-major
__global__ void k_transpose_bf16(const float* __restrict__ W, bf16* __restrict__ Wt,
                                 int K, int Nn) {
    __shared__ float tile[32][33];
    const int k0 = blockIdx.x * 32, n0 = blockIdx.y * 32;
    const int tx = threadIdx.x, ty = threadIdx.y;      // block (32, 8)
    for (int i = ty; i < 32; i += 8)
        tile[i][tx] = W[(long)(k0 + i) * Nn + n0 + tx];
    __syncthreads();
    for (int i = ty; i < 32; i += 8)
        Wt[(long)(n0 + i) * K + k0 + tx] = (bf16)tile[tx][i];
}

// ---------------------------------------------------------------- bf16 WMMA GEMM (double-buffered LDS)
// C[M x Nn] = A[M x K](bf16) * Wt[Nn x K](bf16)^T + bias[Nn] (+ res[M x Nn])
template <bool HAS_RES>
__global__ void __launch_bounds__(256)
k_gemm_bf16(const bf16* __restrict__ A, const bf16* __restrict__ Bt,
            const float* __restrict__ bias, const float* __restrict__ res,
            float* __restrict__ C, int M, int Nn, int K) {
    __shared__ __align__(16) bf16 As[2][128 * 40];
    __shared__ __align__(16) bf16 Bs[2][128 * 40];

    const int tid  = threadIdx.x;
    const int lane = tid & 31, wid = tid >> 5;
    const int wm = wid >> 2, wn = wid & 3;     // 2 x 4 wave grid
    const int l16 = lane & 15, hi = lane >> 4;
    const int bm = blockIdx.y * 128, bn = blockIdx.x * 128;

    v8f acc[4][2];
    const v8f vzero = {0.f,0.f,0.f,0.f,0.f,0.f,0.f,0.f};
    #pragma unroll
    for (int mt = 0; mt < 4; ++mt)
        #pragma unroll
        for (int nt = 0; nt < 2; ++nt) acc[mt][nt] = vzero;

    // staging: each thread moves 16 halves of A and of Bt per K-step
    const int sr = tid >> 1;          // 0..127
    const int sc = (tid & 1) * 16;    // 0 or 16
    const bf16* Aptr = A  + (long)(bm + sr) * K + sc;
    const bf16* Bptr = Bt + (long)(bn + sr) * K + sc;

    auto stage = [&](int buf, int k0) {
        cp8(&As[buf][sr * 40 + sc],     Aptr + k0);
        cp8(&As[buf][sr * 40 + sc + 8], Aptr + k0 + 8);
        cp8(&Bs[buf][sr * 40 + sc],     Bptr + k0);
        cp8(&Bs[buf][sr * 40 + sc + 8], Bptr + k0 + 8);
    };

    stage(0, 0);
    wait_async();
    __syncthreads();

    for (int k0 = 0; k0 < K; k0 += 32) {
        const int cur = (k0 >> 5) & 1;
        if (k0 + 32 < K) stage(cur ^ 1, k0 + 32);   // overlap fill with compute

        Frag aF[4], bF[2];
        const int ksel = hi * 8;
        #pragma unroll
        for (int mt = 0; mt < 4; ++mt) {
            const int m = wm * 64 + mt * 16 + l16;
            aF[mt].h[0] = *(const v8bf*)&As[cur][m * 40 + ksel];
            aF[mt].h[1] = *(const v8bf*)&As[cur][m * 40 + 16 + ksel];
        }
        #pragma unroll
        for (int nt = 0; nt < 2; ++nt) {
            const int n = wn * 32 + nt * 16 + l16;
            bF[nt].h[0] = *(const v8bf*)&Bs[cur][n * 40 + ksel];
            bF[nt].h[1] = *(const v8bf*)&Bs[cur][n * 40 + 16 + ksel];
        }
        #pragma unroll
        for (int mt = 0; mt < 4; ++mt)
            #pragma unroll
            for (int nt = 0; nt < 2; ++nt)
                acc[mt][nt] = wmma_bf16(aF[mt].v, bF[nt].v, acc[mt][nt]);

        wait_async();
        __syncthreads();
    }

    #pragma unroll
    for (int mt = 0; mt < 4; ++mt) {
        #pragma unroll
        for (int nt = 0; nt < 2; ++nt) {
            const int col = bn + wn * 32 + nt * 16 + l16;
            const float bv = bias[col];
            #pragma unroll
            for (int r = 0; r < 8; ++r) {
                const int row = bm + wm * 64 + mt * 16 + r + 8 * hi;
                float vv = acc[mt][nt][r] + bv;
                if (HAS_RES) vv += res[(long)row * Nn + col];
                C[(long)row * Nn + col] = vv;
            }
        }
    }
}

// ---------------------------------------------------------------- adaLN: h = ln(x)*(1+scale)+shift  (bf16 out)
__global__ void __launch_bounds__(256)
k_adaln(const float* __restrict__ x, const float* __restrict__ Smod, bf16* __restrict__ h) {
    const int row = blockIdx.x;
    const float* xr = x + (long)row * DD;
    const float* sr = Smod + (long)row * (2 * DD);
    __shared__ float red0[8], red1[8], stat[2];
    float s = 0.f, s2 = 0.f;
    for (int i = threadIdx.x; i < DD; i += 256) { float v = xr[i]; s += v; s2 += v * v; }
    for (int off = 16; off; off >>= 1) { s += __shfl_xor(s, off); s2 += __shfl_xor(s2, off); }
    const int wid = threadIdx.x >> 5, lane = threadIdx.x & 31;
    if (lane == 0) { red0[wid] = s; red1[wid] = s2; }
    __syncthreads();
    if (threadIdx.x == 0) {
        float a = 0.f, b = 0.f;
        for (int i = 0; i < 8; ++i) { a += red0[i]; b += red1[i]; }
        const float mean = a / DD;
        stat[0] = mean;
        stat[1] = rsqrtf(b / DD - mean * mean + 1e-5f);
    }
    __syncthreads();
    const float mean = stat[0], rstd = stat[1];
    for (int i = threadIdx.x; i < DD; i += 256) {
        const float v = (xr[i] - mean) * rstd;
        h[(long)row * DD + i] = (bf16)(v * (1.f + sr[i]) + sr[DD + i]);
    }
}

// ---------------------------------------------------------------- qkv post: per-head LN(q,k), V transpose, silu(mlp)
__global__ void __launch_bounds__(512)
k_qkvpost(const float* __restrict__ fused,
          bf16* __restrict__ Qn, bf16* __restrict__ Kn,
          bf16* __restrict__ Vt, bf16* __restrict__ Smlp) {
    const int row = blockIdx.x;            // b*N + n
    const int b = row / NN, n = row % NN;
    const float* fr = fused + (long)row * (7 * DD);
    const int tid = threadIdx.x, lane = tid & 31, wid = tid >> 5;   // wid = head

    #pragma unroll
    for (int which = 0; which < 2; ++which) {           // 0: q, 1: k
        const float* src = fr + which * DD + wid * DHH;
        float vals[4], s = 0.f, s2 = 0.f;
        #pragma unroll
        for (int i = 0; i < 4; ++i) { float v = src[lane + 32 * i]; vals[i] = v; s += v; s2 += v * v; }
        for (int off = 16; off; off >>= 1) { s += __shfl_xor(s, off); s2 += __shfl_xor(s2, off); }
        const float mean = s / DHH;
        const float rstd = rsqrtf(s2 / DHH - mean * mean + 1e-5f);
        bf16* dst = (which ? Kn : Qn) + (((long)(b * HH + wid)) * NN + n) * DHH;
        #pragma unroll
        for (int i = 0; i < 4; ++i) dst[lane + 32 * i] = (bf16)((vals[i] - mean) * rstd);
    }
    for (int j = tid; j < DD; j += 512) {               // v, transposed per head: Vt[b][h][dh][n]
        const int hh = j / DHH, dd = j % DHH;
        Vt[(((long)(b * HH + hh)) * DHH + dd) * NN + n] = (bf16)fr[2 * DD + j];
    }
    for (int j = tid; j < 4 * DD; j += 512) {           // silu(mlp_h)
        const float v = fr[3 * DD + j];
        Smlp[(long)row * (4 * DD) + j] = (bf16)(v / (1.f + __expf(-v)));
    }
}

// ---------------------------------------------------------------- flash attention (bf16 WMMA, online softmax)
// Block: 256 threads = 8 waves, each wave owns 16 q-rows; K/V tiles staged
// in LDS once per block (shared by all 8 waves), double-buffered async.
__global__ void __launch_bounds__(256)
k_attention(const bf16* __restrict__ Qn, const bf16* __restrict__ Kn,
            const bf16* __restrict__ Vt, bf16* __restrict__ O) {
    __shared__ __align__(16) bf16 Ks[2][32 * 136];      // [key][dh], padded
    __shared__ __align__(16) bf16 Vs[2][128 * 40];      // [dh][key], padded
    __shared__ __align__(16) bf16 Pl[8][16 * 40];       // per-wave 16x32 P tile

    const int tid = threadIdx.x, lane = tid & 31, wid = tid >> 5;
    const int l16 = lane & 15, hi = lane >> 4;
    const int b = blockIdx.z, h = blockIdx.y;
    const int q0 = blockIdx.x * 128 + wid * 16;         // 8 waves x 16 q-rows

    const bf16* q  = Qn + ((long)(b * HH + h)) * NN * DHH;
    const bf16* kb = Kn + ((long)(b * HH + h)) * NN * DHH;
    const bf16* vb = Vt + ((long)(b * HH + h)) * DHH * NN;

    // K/V tile staging: 2 x 16B per thread per tile
    auto stage = [&](int buf, int j0) {
        #pragma unroll
        for (int j = 0; j < 2; ++j) {
            const int idx = tid * 8 + 2048 * j;
            const int kr = idx >> 7, kc = idx & 127;    // K tile: 32 x 128
            cp8(&Ks[buf][kr * 136 + kc], kb + (long)(j0 + kr) * DHH + kc);
            const int vr = idx >> 5, vc = idx & 31;     // V tile: 128 x 32
            cp8(&Vs[buf][vr * 40 + vc], vb + (long)vr * NN + j0 + vc);
        }
    };

    const int ksel = 8 * hi;
    Frag aQ[4];
    const long mrow = q0 + l16;
    #pragma unroll
    for (int c = 0; c < 4; ++c) {
        aQ[c].h[0] = *(const v8bf*)(q + mrow * DHH + 32 * c + ksel);
        aQ[c].h[1] = *(const v8bf*)(q + mrow * DHH + 32 * c + 16 + ksel);
    }

    const v8f vzero = {0.f,0.f,0.f,0.f,0.f,0.f,0.f,0.f};
    v8f acc[8];
    float mI[8], lI[8];
    #pragma unroll
    for (int t = 0; t < 8; ++t) acc[t] = vzero;
    #pragma unroll
    for (int r = 0; r < 8; ++r) { mI[r] = -1e30f; lI[r] = 0.f; }
    const float sc = 0.08838834764831845f;              // 1/sqrt(128)

    stage(0, 0);
    wait_async();
    __syncthreads();

    for (int j0 = 0; j0 < NN; j0 += 32) {
        const int cur = (j0 >> 5) & 1;
        if (j0 + 32 < NN) stage(cur ^ 1, j0 + 32);      // overlap fill with compute

        v8f s[2] = {vzero, vzero};
        #pragma unroll
        for (int t = 0; t < 2; ++t) {
            const bf16* kr = &Ks[cur][(16 * t + l16) * 136];
            #pragma unroll
            for (int c = 0; c < 4; ++c) {
                Frag bK;
                bK.h[0] = *(const v8bf*)(kr + 32 * c + ksel);
                bK.h[1] = *(const v8bf*)(kr + 32 * c + 16 + ksel);
                s[t] = wmma_bf16(aQ[c].v, bK.v, s[t]);
            }
        }
        #pragma unroll
        for (int r = 0; r < 8; ++r) {
            const float a0 = s[0][r] * sc, a1 = s[1][r] * sc;
            float mx = fmaxf(a0, a1);
            for (int off = 8; off; off >>= 1) mx = fmaxf(mx, __shfl_xor(mx, off));
            const float mN   = fmaxf(mI[r], mx);
            const float corr = __expf(mI[r] - mN);
            const float p0 = __expf(a0 - mN), p1 = __expf(a1 - mN);
            float rs = p0 + p1;
            for (int off = 8; off; off >>= 1) rs += __shfl_xor(rs, off);
            lI[r] = lI[r] * corr + rs;
            mI[r] = mN;
            #pragma unroll
            for (int t = 0; t < 8; ++t) acc[t][r] *= corr;
            Pl[wid][(r + 8 * hi) * 40 + l16]      = (bf16)p0;   // C-layout -> row-major
            Pl[wid][(r + 8 * hi) * 40 + 16 + l16] = (bf16)p1;
        }
        Frag aP;                                        // reload in A-layout
        aP.h[0] = *(const v8bf*)&Pl[wid][l16 * 40 + ksel];
        aP.h[1] = *(const v8bf*)&Pl[wid][l16 * 40 + 16 + ksel];
        #pragma unroll
        for (int t = 0; t < 8; ++t) {
            const bf16* vr = &Vs[cur][(16 * t + l16) * 40 + ksel];
            Frag bV;
            bV.h[0] = *(const v8bf*)(vr);
            bV.h[1] = *(const v8bf*)(vr + 16);
            acc[t] = wmma_bf16(aP.v, bV.v, acc[t]);
        }

        wait_async();
        __syncthreads();
    }
    #pragma unroll
    for (int t = 0; t < 8; ++t)
        #pragma unroll
        for (int r = 0; r < 8; ++r) {
            const int row = q0 + r + 8 * hi;
            O[((long)(b * NN + row)) * DD + h * DHH + 16 * t + l16] = (bf16)(acc[t][r] / lI[r]);
        }
}

// ---------------------------------------------------------------- launch
extern "C" void kernel_launch(void* const* d_in, const int* in_sizes, int n_in,
                              void* d_out, int out_size, void* d_ws, size_t ws_size,
                              hipStream_t stream) {
    const float* x          = (const float*)d_in[0];
    const float* emb        = (const float*)d_in[1];
    const float* emb_w      = (const float*)d_in[2];
    const float* emb_b      = (const float*)d_in[3];
    const float* fused_w    = (const float*)d_in[4];
    const float* fused_b    = (const float*)d_in[5];
    const float* attn_out_w = (const float*)d_in[6];
    const float* attn_out_b = (const float*)d_in[7];
    const float* mlp_w      = (const float*)d_in[8];
    const float* mlp_b      = (const float*)d_in[9];
    float* out = (float*)d_out;

    size_t off = 0;
    auto alloc = [&](size_t bytes) -> void* {
        off = (off + 255) & ~(size_t)255;
        void* p = (char*)d_ws + off;
        off += bytes;
        return p;
    };
    bf16*  emb_bf = (bf16*)alloc((size_t)MM * EE * 2);
    bf16*  ewT    = (bf16*)alloc((size_t)(2 * DD) * EE * 2);
    bf16*  fwT    = (bf16*)alloc((size_t)(7 * DD) * DD * 2);
    bf16*  awT    = (bf16*)alloc((size_t)DD * DD * 2);
    bf16*  mwT    = (bf16*)alloc((size_t)DD * (4 * DD) * 2);
    float* Smod   = (float*)alloc((size_t)MM * (2 * DD) * 4);
    bf16*  hbf    = (bf16*)alloc((size_t)MM * DD * 2);
    float* fusedF = (float*)alloc((size_t)MM * (7 * DD) * 4);
    bf16*  Qn     = (bf16*)alloc((size_t)MM * DD * 2);
    bf16*  Kn     = (bf16*)alloc((size_t)MM * DD * 2);
    bf16*  Vt     = (bf16*)alloc((size_t)MM * DD * 2);
    bf16*  obf    = (bf16*)alloc((size_t)MM * DD * 2);
    bf16*  siluB  = (bf16*)alloc((size_t)MM * (4 * DD) * 2);
    float* xmid   = (float*)alloc((size_t)MM * DD * 4);

    // 1. casts / weight transposes
    {
        long n = (long)MM * EE;
        k_cast_bf16<<<(n + 255) / 256, 256, 0, stream>>>(emb, emb_bf, n);
        dim3 tb(32, 8);
        k_transpose_bf16<<<dim3(EE / 32,       (2 * DD) / 32), tb, 0, stream>>>(emb_w,      ewT, EE,     2 * DD);
        k_transpose_bf16<<<dim3(DD / 32,       (7 * DD) / 32), tb, 0, stream>>>(fused_w,    fwT, DD,     7 * DD);
        k_transpose_bf16<<<dim3(DD / 32,       DD / 32),       tb, 0, stream>>>(attn_out_w, awT, DD,     DD);
        k_transpose_bf16<<<dim3((4 * DD) / 32, DD / 32),       tb, 0, stream>>>(mlp_w,      mwT, 4 * DD, DD);
    }
    // 2. modulation GEMM: Smod = emb @ emb_w + emb_b
    k_gemm_bf16<false><<<dim3((2 * DD) / 128, MM / 128), 256, 0, stream>>>(
        emb_bf, ewT, emb_b, nullptr, Smod, MM, 2 * DD, EE);
    // 3. adaLN
    k_adaln<<<MM, 256, 0, stream>>>(x, Smod, hbf);
    // 4. fused GEMM: fused = h @ fused_w + fused_b
    k_gemm_bf16<false><<<dim3((7 * DD) / 128, MM / 128), 256, 0, stream>>>(
        hbf, fwT, fused_b, nullptr, fusedF, MM, 7 * DD, DD);
    // 5. qkv post-processing
    k_qkvpost<<<MM, 512, 0, stream>>>(fusedF, Qn, Kn, Vt, siluB);
    // 6. flash attention
    k_attention<<<dim3(NN / 128, HH, BB), 256, 0, stream>>>(Qn, Kn, Vt, obf);
    // 7. attn-out GEMM with residual x
    k_gemm_bf16<true><<<dim3(DD / 128, MM / 128), 256, 0, stream>>>(
        obf, awT, attn_out_b, x, xmid, MM, DD, DD);
    // 8. MLP GEMM with residual x_mid -> out
    k_gemm_bf16<true><<<dim3(DD / 128, MM / 128), 256, 0, stream>>>(
        siluB, mwT, mlp_b, xmid, out, MM, DD, 4 * DD);
}
